// IntegralTransform_27891517620878
// MI455X (gfx1250) — matrix operand order repaired and checked by reading, not compiled, not run on hardware
//
#include <hip/hip_runtime.h>
#include <cmath>

typedef __attribute__((ext_vector_type(16))) _Float16 v16h;
typedef __attribute__((ext_vector_type(2)))  _Float16 h2;
typedef __attribute__((ext_vector_type(8)))  float    v8f;

#define DF 64
#define WAVES_PER_BLOCK 4
#define TPB (WAVES_PER_BLOCK * 32)

// ---------------------------------------------------------------------------
// Branchless GELU using gfx1250 hardware tanh (TRANS32 op).
// tanh-form GELU: 0.5*x*(1 + tanh(sqrt(2/pi)*(x + 0.044715*x^3))).
// v_nop after v_tanh_f32 satisfies the TRANS co-execution hazard rule
// (1 independent op before the result may be consumed).
// ---------------------------------------------------------------------------
__device__ __forceinline__ float gelu_hw(float x) {
  float arg = 0.7978845608028654f * (x + 0.044715f * x * x * x);
  float t;
  asm volatile("v_tanh_f32 %0, %1\n\tv_nop" : "=v"(t) : "v"(arg));
  return 0.5f * x * (1.0f + t);
}

// ---------------------------------------------------------------------------
// Zero the output accumulator (harness poisons d_out with 0xAA).
// ---------------------------------------------------------------------------
__global__ void it_zero_kernel(float* __restrict__ p, int n) {
  int i = blockIdx.x * blockDim.x + threadIdx.x;
  if (i < n) p[i] = 0.0f;
}

// ---------------------------------------------------------------------------
// CSR row_splits -> per-edge segment id (inverse of repeat_interleave).
// One thread per node fills its edge range (avg 16 edges/node).
// ---------------------------------------------------------------------------
__global__ void it_seg_kernel(const long long* __restrict__ splits,
                              int* __restrict__ seg, int n) {
  int i = blockIdx.x * blockDim.x + threadIdx.x;
  if (i < n) {
    long long b = splits[i], e = splits[i + 1];
    for (long long j = b; j < e; ++j) seg[j] = i;
  }
}

// ---------------------------------------------------------------------------
// Divide segment sums by max(count,1) -> segment mean.
// ---------------------------------------------------------------------------
__global__ void it_norm_kernel(float* __restrict__ out,
                               const long long* __restrict__ splits,
                               int total) {
  int i = blockIdx.x * blockDim.x + threadIdx.x;
  if (i < total) {
    int row = i >> 6;  // /DF
    float cnt = (float)(splits[row + 1] - splits[row]);
    out[i] = out[i] / fmaxf(cnt, 1.0f);
  }
}

// ---------------------------------------------------------------------------
// Main edge kernel: one wave processes a tile of 16 edges.
//   agg[16x6] --wmma--> gelu --LDS transpose--> H[16x64] --wmma--> D[16x64]
//   D *= f_y[nbr]; atomic-accumulate into out[seg].
// ---------------------------------------------------------------------------
__global__ __launch_bounds__(TPB)
void it_edge_kernel(const float* __restrict__ y,
                    const float* __restrict__ f_y,
                    const float* __restrict__ W0,
                    const float* __restrict__ b0,
                    const float* __restrict__ W1,
                    const float* __restrict__ b1,
                    const long long* __restrict__ nbr,
                    const int* __restrict__ seg,
                    float* __restrict__ out,
                    int ntiles, long long E) {
  __shared__ _Float16 stage_all[WAVES_PER_BLOCK][16 * DF];  // 2KB per wave

  const int lane = threadIdx.x & 31;
  const int wib  = threadIdx.x >> 5;
  const int half = lane >> 4;   // lane half: selects K-subgroup (A/B layouts)
  const int lcol = lane & 15;   // column within a 16-wide tile (B/C/D layouts)
  _Float16* stage = &stage_all[wib][0];

  // ---- Hoist weights into registers in WMMA B-matrix layout (f16) ----
  // B operand in this file's k order, shared with A (16-bit, KxN=32x16): lane L holds column N=L%16; element 2r+q
  // corresponds to K = (L/16)*16 + 2r + q.
  v16h B0[4];        // W0 padded [32 x 64], per 16-column tile
  v16h B1m[4][2];    // W1 [64 x 64], per (column tile, K-half)
  float b0c[4], b1c[4];
#pragma unroll
  for (int n = 0; n < 4; ++n) {
    const int c = n * 16 + lcol;
    b0c[n] = b0[c];
    b1c[n] = b1[c];
#pragma unroll
    for (int r = 0; r < 8; ++r) {
#pragma unroll
      for (int q = 0; q < 2; ++q) {
        const int k = half * 16 + 2 * r + q;
        B0[n][2 * r + q] = (k < 6) ? (_Float16)W0[k * DF + c] : (_Float16)0.0f;
#pragma unroll
        for (int w = 0; w < 2; ++w)
          B1m[n][w][2 * r + q] = (_Float16)W1[(w * 32 + k) * DF + c];
      }
    }
  }

  const int nwaves = (int)gridDim.x * WAVES_PER_BLOCK;
  const int wgid   = (int)blockIdx.x * WAVES_PER_BLOCK + wib;

  for (int tile = wgid; tile < ntiles; tile += nwaves) {
    const long long ebase = (long long)tile * 16;

    // Prefetch next tile's index stream (speculative, no counter cost).
    __builtin_prefetch(nbr + (long long)(tile + nwaves) * 16, 0, 1);

    // ---- Build layer-0 A operand: agg[16x6] zero-padded to 16x32 f16 ----
    // A layout (16-bit, MxK=16x32): lane L holds row M=L%16. For half==0 the
    // first 6 elements map to K=0..5 == agg[0..5]. Lanes of half==1 write the
    // same loaded values into K=16..21 — harmless, since B0 is zero there
    // (only finiteness matters: WMMA maps 0*Inf -> NaN). Fully branchless:
    // both halves compute identical addresses, hardware merges the loads.
    long long eSelf = ebase + lcol;
    eSelf = (eSelf < E) ? eSelf : (E - 1);  // tail clamp, no divergence
    const long long nb = nbr[eSelf];
    const long long sg = seg[eSelf];
    v16h A0 = {};
    A0[0] = (_Float16)y[nb * 3 + 0];
    A0[1] = (_Float16)y[nb * 3 + 1];
    A0[2] = (_Float16)y[nb * 3 + 2];
    A0[3] = (_Float16)y[sg * 3 + 0];
    A0[4] = (_Float16)y[sg * 3 + 1];
    A0[5] = (_Float16)y[sg * 3 + 2];

    // ---- Layer 0: Hpre = agg @ W0 + b0 (C initialized to b0 broadcast) ----
    v8f C[4];
#pragma unroll
    for (int n = 0; n < 4; ++n) {
      v8f cc;
#pragma unroll
      for (int i = 0; i < 8; ++i) cc[i] = b0c[n];
      C[n] = __builtin_amdgcn_wmma_f32_16x16x32_f16(
          false, A0, false, B0[n], (short)0, cc, false, false);
    }

    // ---- Hardware-tanh GELU in D layout, stash transposed to LDS [m][k] ----
    // D layout: lane L holds column c = n*16 + L%16, rows m = r + 8*(L/16).
#pragma unroll
    for (int n = 0; n < 4; ++n) {
      const int c = n * 16 + lcol;
#pragma unroll
      for (int r = 0; r < 8; ++r) {
        const int m = r + 8 * half;
        stage[m * DF + c] = (_Float16)gelu_hw(C[n][r]);
      }
    }
    // Per-wave LDS RAW fence (waves have different trip counts; no block
    // barrier is legal here). LDS ops are in-order per wave; this drains them.
    asm volatile("s_wait_dscnt 0x0" ::: "memory");

    // ---- Reload H in A-matrix layout (packed f16 pairs, K even).
    // Within each r-quad the pair addresses are contiguous, so the compiler
    // merges these into ds_load_b128s.
    v16h A1[2];
#pragma unroll
    for (int w = 0; w < 2; ++w) {
#pragma unroll
      for (int r = 0; r < 8; ++r) {
        const int k = w * 32 + half * 16 + r * 2;
        const h2 p = *(const h2*)(stage + lcol * DF + k);
        A1[w][2 * r + 0] = p[0];
        A1[w][2 * r + 1] = p[1];
      }
    }

    // ---- Layer 1: D = H @ W1 + b1 (K=64 via two chained WMMAs) ----
    v8f Dt[4];
#pragma unroll
    for (int n = 0; n < 4; ++n) {
      v8f cc;
#pragma unroll
      for (int i = 0; i < 8; ++i) cc[i] = b1c[n];
      cc = __builtin_amdgcn_wmma_f32_16x16x32_f16(
          false, A1[0], false, B1m[n][0], (short)0, cc, false, false);
      cc = __builtin_amdgcn_wmma_f32_16x16x32_f16(
          false, A1[1], false, B1m[n][1], (short)0, cc, false, false);
      Dt[n] = cc;
    }

    // ---- Modulate by f_y[nbr] and scatter-add into out[seg] ----
    // Per (r): 16 lanes of each half read/write 16 consecutive floats of one
    // row -> each VMEM op touches two contiguous 64B spans (coalesced).
#pragma unroll
    for (int r = 0; r < 8; ++r) {
      const int m = r + 8 * half;
      const long long em = ebase + m;
      if (em < E) {
        const long long nbm = nbr[em];
        const long long sgm = seg[em];
#pragma unroll
        for (int n = 0; n < 4; ++n) {
          const int c = n * 16 + lcol;
          atomicAdd(out + sgm * DF + c, Dt[n][r] * f_y[nbm * DF + c]);
        }
      }
    }
    // Next iteration's stage stores cannot clobber in-flight A1 loads:
    // LDS instructions from one wave are processed in order.
  }
}

// ---------------------------------------------------------------------------
// Launch
// ---------------------------------------------------------------------------
extern "C" void kernel_launch(void* const* d_in, const int* in_sizes, int n_in,
                              void* d_out, int out_size, void* d_ws, size_t ws_size,
                              hipStream_t stream) {
  const float*     y      = (const float*)d_in[0];
  const float*     f_y    = (const float*)d_in[1];
  const float*     W0     = (const float*)d_in[2];
  const float*     b0     = (const float*)d_in[3];
  const float*     W1     = (const float*)d_in[4];
  const float*     b1     = (const float*)d_in[5];
  const long long* nbr    = (const long long*)d_in[6];
  const long long* splits = (const long long*)d_in[7];
  float* out = (float*)d_out;

  const int       N = in_sizes[0] / 3;
  const long long E = (long long)in_sizes[6];
  const int   total = N * DF;
  int* seg = (int*)d_ws;  // E ints of workspace

  it_zero_kernel<<<(total + 255) / 256, 256, 0, stream>>>(out, total);
  it_seg_kernel<<<(N + 255) / 256, 256, 0, stream>>>(splits, seg, N);

  const int ntiles = (int)((E + 15) / 16);
  int blocks = (ntiles + WAVES_PER_BLOCK - 1) / WAVES_PER_BLOCK;
  if (blocks > 4096) blocks = 4096;
  if (blocks < 1) blocks = 1;
  it_edge_kernel<<<blocks, TPB, 0, stream>>>(y, f_y, W0, b0, W1, b1,
                                             nbr, seg, out, ntiles, E);

  it_norm_kernel<<<(total + 255) / 256, 256, 0, stream>>>(out, splits, total);
}
